// IntraDomainAttention_54743653155456
// MI455X (gfx1250) — compile-verified, matrix operand order checked by reading, not run on hardware
//
#include <hip/hip_runtime.h>
#include <hip/hip_bf16.h>

// MI455X / gfx1250, wave32. Matrix ops via v_wmma_f32_16x16x32_f16.

typedef __attribute__((ext_vector_type(16))) _Float16 v16h;
typedef __attribute__((ext_vector_type(8)))  _Float16 h8;
typedef __attribute__((ext_vector_type(8)))  float    v8f;
typedef __attribute__((ext_vector_type(4)))  float    f4;

#define WMMA16(a, b, c) __builtin_amdgcn_wmma_f32_16x16x32_f16( \
    false, (a), false, (b), (short)0, (c), false, false)

constexpr int Bb = 8, Nn = 2048, Mm = 1024, Cc = 512, Hh = 8, Dd = 64;

__device__ __forceinline__ v16h pack16(h8 lo, h8 hi) {
    v16h r;
#pragma unroll
    for (int i = 0; i < 8; ++i) { r[i] = lo[i]; r[8 + i] = hi[i]; }
    return r;
}

// reductions within each 16-lane half (xor masks < 16 never cross bit 4)
__device__ __forceinline__ float hmax16(float v) {
#pragma unroll
    for (int m = 1; m < 16; m <<= 1) v = fmaxf(v, __shfl_xor(v, m, 32));
    return v;
}
__device__ __forceinline__ float hsum16(float v) {
#pragma unroll
    for (int m = 1; m < 16; m <<= 1) v += __shfl_xor(v, m, 32);
    return v;
}

// ---------------------------------------------------------------------------
// Projection GEMM:  out = A[rows x 512] * W[512 x cols] + bias  (f32 in, f16 out)
// Block = 128 threads (4 waves). Each wave: one 16x16 output tile.
// mode 0: store Q as [B,H,N,64].  mode 1: cols 0..511 -> K [B,H,M,64],
//                                 cols 512..1023 -> V^T [B,H,64,M].
// ---------------------------------------------------------------------------
__global__ void __launch_bounds__(128)
proj_gemm_kernel(const float* __restrict__ A, const float* __restrict__ W,
                 const float* __restrict__ bias, int ldw, int mode,
                 _Float16* __restrict__ out0, _Float16* __restrict__ out1) {
    __shared__ alignas(16) _Float16 ldsWt[64][32];  // [col in block][k], 4 KB

    const int wave = threadIdx.x >> 5;
    const int lane = threadIdx.x & 31;
    const int half = lane >> 4;
    const int ln   = lane & 15;
    const long rowTile = (long)blockIdx.y * 16;
    const long colBase = (long)blockIdx.x * 64;
    const long rowA    = rowTile + ln;

    v8f acc = {};

    for (int k0 = 0; k0 < Cc; k0 += 32) {
        // stage 32x64 chunk of W into LDS, transposed, as f16 (coalesced reads)
#pragma unroll
        for (int i = 0; i < 16; ++i) {
            int idx = threadIdx.x + i * 128;
            int kk  = idx >> 6;
            int cc  = idx & 63;
            ldsWt[cc][kk] = (_Float16)W[(long)(k0 + kk) * ldw + colBase + cc];
        }
        __syncthreads();

        const float* ap = A + rowA * Cc + k0;
        f4 x0 = *(const f4*)(ap + half * 8);
        f4 x1 = *(const f4*)(ap + half * 8 + 4);
        f4 x2 = *(const f4*)(ap + 16 + half * 8);
        f4 x3 = *(const f4*)(ap + 16 + half * 8 + 4);
        v16h a;
#pragma unroll
        for (int i = 0; i < 4; ++i) {
            a[i]      = (_Float16)x0[i];
            a[4 + i]  = (_Float16)x1[i];
            a[8 + i]  = (_Float16)x2[i];
            a[12 + i] = (_Float16)x3[i];
        }

        const h8* wp = (const h8*)&ldsWt[wave * 16 + ln][0];
        v16h b = pack16(wp[half], wp[2 + half]);

        acc = WMMA16(a, b, acc);
        __syncthreads();
    }

    const int colOut = (int)colBase + wave * 16 + ln;
    const float bv = bias[colOut];

#pragma unroll
    for (int r = 0; r < 8; ++r) {
        const long row = rowTile + half * 8 + r;
        const _Float16 val = (_Float16)(acc[r] + bv);
        if (mode == 0) {
            const long b = row >> 11, n = row & 2047;
            const int  h = colOut >> 6, d = colOut & 63;
            out0[(((b * Hh + h) * (long)Nn) + n) * Dd + d] = val;
        } else {
            const long b = row >> 10, m = row & 1023;
            if (colOut < Cc) {
                const int h = colOut >> 6, d = colOut & 63;
                out0[(((b * Hh + h) * (long)Mm) + m) * Dd + d] = val;
            } else {
                const int cc2 = colOut - Cc;
                const int h = cc2 >> 6, d = cc2 & 63;
                out1[(((b * Hh + h) * (long)Dd) + d) * Mm + m] = val;
            }
        }
    }
}

// ---------------------------------------------------------------------------
// Flash attention: block = 256 threads (8 waves), each wave owns 16 q rows.
// K/V chunks (32 keys) staged to LDS ONCE per block (8x less L2 traffic),
// double-buffered and software-pipelined: next chunk's global loads issue
// before compute, LDS stores land after compute, one barrier per chunk.
// ---------------------------------------------------------------------------
__global__ void __launch_bounds__(256)
attn_kernel(const _Float16* __restrict__ Q, const _Float16* __restrict__ Kf,
            const _Float16* __restrict__ Vt, float* __restrict__ out) {
    __shared__ alignas(16) _Float16 ldsK[2][32][64];   // keys x d, 8 KB
    __shared__ alignas(16) _Float16 ldsV[2][64][32];   // d x m,    8 KB
    __shared__ alignas(16) _Float16 ldsP[8][16][32];   // per-wave P strip, 8 KB

    const int tid  = threadIdx.x;
    const int wave = tid >> 5;
    const int lane = tid & 31;
    const int half = lane >> 4;
    const int ln   = lane & 15;

    const int bh = blockIdx.x;                 // 0..63
    const int b  = bh >> 3, h = bh & 7;
    const int n0 = blockIdx.y * 128 + wave * 16;

    const _Float16* qbase = Q  + ((long)bh * Nn + n0) * Dd;
    const _Float16* kbase = Kf + (long)bh * Mm * Dd;
    const _Float16* vbase = Vt + (long)bh * Dd * Mm;

    // cooperative-staging coordinates (one b128 per thread per array)
    const int krow = tid >> 3, kseg = tid & 7;         // K: 32 rows x 8 segs
    const int vrow = tid >> 2, vseg = tid & 3;         // V: 64 rows x 4 segs

    // Q operands held in registers for the whole M loop
    const h8* qp = (const h8*)(qbase + (long)ln * Dd);
    const v16h qa0 = pack16(qp[half],     qp[2 + half]);   // d = 0..31
    const v16h qa1 = pack16(qp[4 + half], qp[6 + half]);   // d = 32..63

    // stage chunk 0
    *(h8*)&ldsK[0][krow][kseg * 8] =
        *(const h8*)(kbase + (long)krow * Dd + kseg * 8);
    *(h8*)&ldsV[0][vrow][vseg * 8] =
        *(const h8*)(vbase + (long)vrow * Mm + vseg * 8);
    __syncthreads();

    v8f o0 = {}, o1 = {}, o2 = {}, o3 = {};
    float mi[8], li[8];
#pragma unroll
    for (int r = 0; r < 8; ++r) { mi[r] = -1e30f; li[r] = 0.f; }

    constexpr int NITER = Mm / 32;             // 32 chunks
    for (int it = 0; it < NITER; ++it) {
        const int buf = it & 1;
        const int m0  = it * 32;

        // ---- issue next chunk's global loads into registers (latency hides
        //      under the WMMA/softmax below); prefetch chunk it+2 into L2.
        h8 kreg, vreg;
        const bool have_next = (it + 1 < NITER);
        if (have_next) {
            kreg = *(const h8*)(kbase + (long)(m0 + 32 + krow) * Dd + kseg * 8);
            vreg = *(const h8*)(vbase + (long)vrow * Mm + (m0 + 32) + vseg * 8);
            if (it + 2 < NITER) {
                __builtin_prefetch(kbase + (long)(m0 + 64 + krow) * Dd + kseg * 8, 0, 1);
                __builtin_prefetch(vbase + (long)vrow * Mm + (m0 + 64) + vseg * 8, 0, 1);
            }
        }

        // ---- scores for keys [m0, m0+16) and [m0+16, m0+32) from LDS
        v8f s0 = {}, s1 = {};
        {
            const h8* kp0 = (const h8*)&ldsK[buf][ln][0];
            s0 = WMMA16(qa0, pack16(kp0[half],     kp0[2 + half]), s0);
            s0 = WMMA16(qa1, pack16(kp0[4 + half], kp0[6 + half]), s0);
            const h8* kp1 = (const h8*)&ldsK[buf][16 + ln][0];
            s1 = WMMA16(qa0, pack16(kp1[half],     kp1[2 + half]), s1);
            s1 = WMMA16(qa1, pack16(kp1[4 + half], kp1[6 + half]), s1);
        }

        // ---- online softmax (rows live across 16-lane halves)
        float scale[8];
#pragma unroll
        for (int r = 0; r < 8; ++r) {
            float t = hmax16(fmaxf(s0[r], s1[r]));
            float mnew = fmaxf(mi[r], t);
            scale[r] = __expf(mi[r] - mnew);
            float p0 = __expf(s0[r] - mnew);
            float p1 = __expf(s1[r] - mnew);
            s0[r] = p0; s1[r] = p1;
            li[r] = li[r] * scale[r] + hsum16(p0 + p1);
            mi[r] = mnew;
        }
#pragma unroll
        for (int r = 0; r < 8; ++r) {
            o0[r] *= scale[r]; o1[r] *= scale[r];
            o2[r] *= scale[r]; o3[r] *= scale[r];
        }

        // ---- transpose P: C/D layout -> A-operand layout via wave's LDS strip
#pragma unroll
        for (int r = 0; r < 8; ++r) {
            const int row = half * 8 + r;
            ldsP[wave][row][ln]      = (_Float16)s0[r];
            ldsP[wave][row][16 + ln] = (_Float16)s1[r];
        }
        asm volatile("s_wait_dscnt 0x0" ::: "memory");
        __builtin_amdgcn_wave_barrier();

        const h8* pp = (const h8*)&ldsP[wave][ln][0];
        const v16h pA = pack16(pp[half], pp[2 + half]);

        // ---- O += P * V from LDS (V^T rows contiguous along m)
        {
            const h8* vp0 = (const h8*)&ldsV[buf][0 * 16 + ln][0];
            o0 = WMMA16(pA, pack16(vp0[half], vp0[2 + half]), o0);
            const h8* vp1 = (const h8*)&ldsV[buf][1 * 16 + ln][0];
            o1 = WMMA16(pA, pack16(vp1[half], vp1[2 + half]), o1);
            const h8* vp2 = (const h8*)&ldsV[buf][2 * 16 + ln][0];
            o2 = WMMA16(pA, pack16(vp2[half], vp2[2 + half]), o2);
            const h8* vp3 = (const h8*)&ldsV[buf][3 * 16 + ln][0];
            o3 = WMMA16(pA, pack16(vp3[half], vp3[2 + half]), o3);
        }

        // ---- commit next chunk to the alternate LDS buffer
        if (have_next) {
            *(h8*)&ldsK[buf ^ 1][krow][kseg * 8] = kreg;
            *(h8*)&ldsV[buf ^ 1][vrow][vseg * 8] = vreg;
        }
        __syncthreads();
    }

    // ---- normalize and store out[b, n, h*64 + d]
#pragma unroll
    for (int r = 0; r < 8; ++r) {
        const float inv = 1.f / li[r];
        const long nrow = n0 + half * 8 + r;
        float* orow = out + ((long)b * Nn + nrow) * Cc + h * Dd;
        orow[0  + ln] = o0[r] * inv;
        orow[16 + ln] = o1[r] * inv;
        orow[32 + ln] = o2[r] * inv;
        orow[48 + ln] = o3[r] * inv;
    }
}

// ---------------------------------------------------------------------------
extern "C" void kernel_launch(void* const* d_in, const int* in_sizes, int n_in,
                              void* d_out, int out_size, void* d_ws, size_t ws_size,
                              hipStream_t stream) {
    const float* x   = (const float*)d_in[0];
    const float* gt  = (const float*)d_in[1];
    const float* Wq  = (const float*)d_in[2];
    const float* bq  = (const float*)d_in[3];
    const float* Wkv = (const float*)d_in[4];
    const float* bkv = (const float*)d_in[5];
    float* out = (float*)d_out;

    // workspace layout: 16 MB Qf16 + 8 MB Kf16 + 8 MB V^T f16 = 32 MB
    char* ws = (char*)d_ws;
    _Float16* qf = (_Float16*)(ws);
    _Float16* kf = (_Float16*)(ws + (size_t)16 * 1024 * 1024);
    _Float16* vt = (_Float16*)(ws + (size_t)24 * 1024 * 1024);

    // Q projection: 16384 x 512
    dim3 gq(Cc / 64, (Bb * Nn) / 16);          // (8, 1024)
    proj_gemm_kernel<<<gq, 128, 0, stream>>>(x, Wq, bq, Cc, 0, qf, qf);

    // KV projection: 8192 x 1024
    dim3 gkv((2 * Cc) / 64, (Bb * Mm) / 16);   // (16, 512)
    proj_gemm_kernel<<<gkv, 128, 0, stream>>>(gt, Wkv, bkv, 2 * Cc, 1, kf, vt);

    // attention
    dim3 ga(Bb * Hh, Nn / 128);                // (64, 16)
    attn_kernel<<<ga, 256, 0, stream>>>(qf, kf, vt, out);
}
